// Classifier_51393578664334
// MI455X (gfx1250) — compile-verified, matrix operand order checked by reading, not run on hardware
//
#include <hip/hip_runtime.h>
#include <hip/hip_bf16.h>
#include <math.h>

// ---------------- problem constants ----------------
constexpr int CB    = 32;            // batch
constexpr int CL    = 128;           // seq len
constexpr int CE    = 300;           // embed dim
constexpr int CH    = 512;           // hidden
constexpr int CDOUT = 5;
constexpr int CBL   = CB * CL;       // 4096 rows
constexpr int CKG   = 3 * CH + CE + CH;  // 2348 ctx features
constexpr int CKGP  = 2368;          // padded to x32 (74 stages, even)
constexpr int CNG   = 7 * CH;        // 3584 gate outputs
constexpr int CK0P  = 320;           // E padded to x32 (10 stages, even)
constexpr int CKFI  = 2 * CH;        // 1024 (32 stages, even)
constexpr int CNSTEPS = 5;

typedef __attribute__((ext_vector_type(16))) __bf16 v16bf;
typedef __attribute__((ext_vector_type(8)))  float  v8f;

// ---------------- embedding gather + bf16 pack (padded K) ----------------
__global__ void embed_pack_kernel(const int* __restrict__ tokens,
                                  const float* __restrict__ embed,
                                  float* __restrict__ x,
                                  __bf16* __restrict__ xb)
{
    long idx = (long)blockIdx.x * blockDim.x + threadIdx.x;
    if (idx >= (long)CBL * CK0P) return;
    int row = (int)(idx / CK0P);
    int col = (int)(idx % CK0P);
    float v = 0.0f;
    if (col < CE) {
        int tok = tokens[row];
        v = embed[(long)tok * CE + col];
        x[(long)row * CE + col] = v;
    }
    xb[idx] = (__bf16)v;
}

// -------- fp32 W[K,N] -> bf16 W^T[N,Kp] (transposed + K padded to x32) --------
__global__ void convert_pad_T_kernel(const float* __restrict__ src,
                                     __bf16* __restrict__ dst,
                                     int K, int Kp, int N)
{
    long idx = (long)blockIdx.x * blockDim.x + threadIdx.x;
    if (idx >= (long)N * Kp) return;
    int n = (int)(idx / Kp);
    int k = (int)(idx % Kp);
    float v = (k < K) ? src[(long)k * N + n] : 0.0f;
    dst[idx] = (__bf16)v;
}

// ---------------- bf16 WMMA GEMM: C[M,N] = A[M,K] * Bt[N,K]^T + bias ----------------
// Requires: M % 128 == 0, N % 128 == 0, K % 64 == 0 (even number of 32-wide stages).
// Double-buffered LDS staged with CDNA5 async global->LDS copies (ASYNCcnt),
// software pipeline manually unrolled x2 so buffer indices are compile-time constants.
// mode 0: C = acc + bias
// mode 1: C = tanh(acc + bias) * mask(row)   (row -> (b = row>>7, l = row&127))
__global__ __launch_bounds__(256)
void gemm_bf16_wmma_kernel(const __bf16* __restrict__ A,
                           const __bf16* __restrict__ Bt,
                           const float* __restrict__ bias,
                           float* __restrict__ C,
                           int M, int N, int K, int mode,
                           const int* __restrict__ lengths)
{
    constexpr int LDK = 40;  // halves; 80B row stride (16B aligned)
    __shared__ __align__(16) __bf16 As[2][128][LDK];
    __shared__ __align__(16) __bf16 Bs[2][128][LDK];

    const int t    = threadIdx.x;
    const int lane = t & 31;
    const int wave = t >> 5;
    const int bm   = blockIdx.x * 128;
    const int bn   = blockIdx.y * 128;
    const int wm   = (wave >> 2) * 64;   // 2 waves along M
    const int wn   = (wave & 3) * 32;    // 4 waves along N

    // staging coords: each thread copies 32B (two b128 chunks) per tile per matrix
    const int srow = t >> 1;            // 0..127 : tile row
    const int sch  = (t & 1) * 16;      // half-offset within row (0 or 16)
    const unsigned goff0 = (unsigned)((srow * K + sch) * 2);  // byte offset vs block base
    const __bf16* aBase = A  + (size_t)bm * K;
    const __bf16* bBase = Bt + (size_t)bn * K;

    unsigned ldsA[2], ldsB[2];
    ldsA[0] = (unsigned)(size_t)(void*)&As[0][srow][sch];
    ldsA[1] = (unsigned)(size_t)(void*)&As[1][srow][sch];
    ldsB[0] = (unsigned)(size_t)(void*)&Bs[0][srow][sch];
    ldsB[1] = (unsigned)(size_t)(void*)&Bs[1][srow][sch];

    v8f acc[4][2];
    #pragma unroll
    for (int mt = 0; mt < 4; ++mt)
        #pragma unroll
        for (int nt = 0; nt < 2; ++nt)
            acc[mt][nt] = (v8f){0.f,0.f,0.f,0.f,0.f,0.f,0.f,0.f};

    const int kb = (lane >> 4) * 8;   // per-lane K base within the 32-wide stage
    const int ln = lane & 15;
    const int nstage = K >> 5;        // even by contract

    // issue: async copy stage tile (k0) into LDS buffer `buf`; +4 ASYNCcnt per wave
    auto issue = [&](int stage, int buf) {
        unsigned goff = goff0 + ((unsigned)stage << 6);  // 64 bytes per stage
        asm volatile("global_load_async_to_lds_b128 %0, %1, %2\n\t"
                     "global_load_async_to_lds_b128 %0, %1, %2 offset:16"
                     :: "v"(ldsA[buf]), "v"(goff), "s"(aBase) : "memory");
        asm volatile("global_load_async_to_lds_b128 %0, %1, %2\n\t"
                     "global_load_async_to_lds_b128 %0, %1, %2 offset:16"
                     :: "v"(ldsB[buf]), "v"(goff), "s"(bBase) : "memory");
    };

    // compute: 6 fragment loads + 8 WMMAs from LDS buffer `p` (constant at call site)
    auto compute = [&](int p) {
        v16bf bfrag[2];
        #pragma unroll
        for (int nt = 0; nt < 2; ++nt) {
            int n = wn + nt * 16 + ln;
            v16bf bv;
            #pragma unroll
            for (int i = 0; i < 8; ++i) {
                bv[i]     = Bs[p][n][kb + i];
                bv[i + 8] = Bs[p][n][kb + 16 + i];
            }
            bfrag[nt] = bv;
        }
        #pragma unroll
        for (int mt = 0; mt < 4; ++mt) {
            int m = wm + mt * 16 + ln;
            v16bf av;
            #pragma unroll
            for (int i = 0; i < 8; ++i) {
                av[i]     = As[p][m][kb + i];
                av[i + 8] = As[p][m][kb + 16 + i];
            }
            #pragma unroll
            for (int nt = 0; nt < 2; ++nt) {
                acc[mt][nt] = __builtin_amdgcn_wmma_f32_16x16x32_bf16(
                    false, av, false, bfrag[nt], (short)0, acc[mt][nt], false, false);
            }
        }
    };

    issue(0, 0);

    // steady state: two stages per iteration, constant buffer indices
    for (int s = 0; s < nstage - 2; s += 2) {
        issue(s + 1, 1);
        asm volatile("s_wait_asynccnt 0x4" ::: "memory");  // stage s landed (in-order)
        __syncthreads();
        compute(0);
        __syncthreads();                                    // safe to refill buffer 0

        issue(s + 2, 0);
        asm volatile("s_wait_asynccnt 0x4" ::: "memory");  // stage s+1 landed
        __syncthreads();
        compute(1);
        __syncthreads();                                    // safe to refill buffer 1
    }

    // epilogue: stages nstage-2 (already in buffer 0) and nstage-1
    issue(nstage - 1, 1);
    asm volatile("s_wait_asynccnt 0x4" ::: "memory");
    __syncthreads();
    compute(0);
    __syncthreads();

    asm volatile("s_wait_asynccnt 0x0" ::: "memory");
    __syncthreads();
    compute(1);

    // ---- epilogue: C/D layout is M = r + 8*(lane>=16), N = lane&15 ----
    const int lm = (lane >> 4) * 8;
    #pragma unroll
    for (int mt = 0; mt < 4; ++mt) {
        #pragma unroll
        for (int nt = 0; nt < 2; ++nt) {
            int gn = bn + wn + nt * 16 + ln;
            float bvadd = bias[gn];
            #pragma unroll
            for (int r = 0; r < 8; ++r) {
                int gm = bm + wm + mt * 16 + lm + r;
                float v = acc[mt][nt][r] + bvadd;
                if (mode == 1) {
                    int bidx = gm >> 7;
                    int lpos = gm & 127;
                    v = tanhf(v) * ((lpos < lengths[bidx]) ? 1.0f : 0.0f);
                }
                C[(size_t)gm * N + gn] = v;
            }
        }
    }
}

// ---------------- ctx packing: [hl | h | hr | x | g] -> bf16, K padded ----------------
__global__ void pack_ctx_kernel(const float* __restrict__ h,
                                const float* __restrict__ x,
                                const float* __restrict__ g,
                                __bf16* __restrict__ ctx)
{
    long idx = (long)blockIdx.x * blockDim.x + threadIdx.x;
    if (idx >= (long)CBL * CKGP) return;
    int row = (int)(idx / CKGP);
    int col = (int)(idx % CKGP);
    int b   = row >> 7;
    int l   = row & 127;
    float v = 0.0f;
    if (col < CH) {
        if (l > 0) v = h[(long)(row - 1) * CH + col];
    } else if (col < 2 * CH) {
        v = h[(long)row * CH + (col - CH)];
    } else if (col < 3 * CH) {
        if (l < CL - 1) v = h[(long)(row + 1) * CH + (col - 2 * CH)];
    } else if (col < 3 * CH + CE) {
        v = x[(long)row * CE + (col - 3 * CH)];
    } else if (col < CKG) {
        v = g[(long)b * CH + (col - (3 * CH + CE))];
    }
    ctx[idx] = (__bf16)v;
}

// ---------------- gate nonlinearities + cell/hidden update ----------------
__global__ void cell_update_kernel(const float* __restrict__ gates,
                                   const float* __restrict__ c,
                                   const float* __restrict__ cg,
                                   const int* __restrict__ lengths,
                                   float* __restrict__ c_new,
                                   float* __restrict__ h_new)
{
    long idx = (long)blockIdx.x * blockDim.x + threadIdx.x;
    if (idx >= (long)CBL * CH) return;
    int row = (int)(idx / CH);
    int hd  = (int)(idx % CH);
    int b   = row >> 7;
    int l   = row & 127;

    const float* gp = gates + (size_t)row * CNG + hd;
    float gi = gp[0 * CH], gl = gp[1 * CH], gf = gp[2 * CH];
    float gr = gp[3 * CH], gs = gp[4 * CH], go = gp[5 * CH], gu = gp[6 * CH];

    float mx = fmaxf(fmaxf(fmaxf(gi, gl), fmaxf(gf, gr)), gs);
    float ei = expf(gi - mx), el = expf(gl - mx), ef = expf(gf - mx);
    float er = expf(gr - mx), es = expf(gs - mx);
    float inv = 1.0f / (ei + el + ef + er + es);
    float i_ = ei * inv, l_ = el * inv, f_ = ef * inv, r_ = er * inv, s_ = es * inv;

    float o = 1.0f / (1.0f + expf(-go));
    float u = tanhf(gu);

    float cl = (l > 0)      ? c[idx - CH] : 0.0f;
    float cr = (l < CL - 1) ? c[idx + CH] : 0.0f;
    float cn = l_ * cl + f_ * c[idx] + r_ * cr + s_ * cg[(long)b * CH + hd] + i_ * u;

    float msk = (l < lengths[b]) ? 1.0f : 0.0f;
    cn *= msk;
    c_new[idx] = cn;
    h_new[idx] = o * tanhf(cn) * msk;
}

// ---------------- sequence average over L (masked rows already zero) ----------------
__global__ void seq_avg_kernel(const float* __restrict__ hin,
                               const int* __restrict__ lengths,
                               float* __restrict__ out)
{
    long idx = (long)blockIdx.x * blockDim.x + threadIdx.x;
    if (idx >= (long)CB * CH) return;
    int b  = (int)(idx / CH);
    int hd = (int)(idx % CH);
    float s = 0.0f;
    for (int l = 0; l < CL; ++l)
        s += hin[(long)(b * CL + l) * CH + hd];
    float len = fmaxf((float)lengths[b], 1.0f);
    out[idx] = s / len;
}

// ---------------- global-node gates: fg, og from concat(g, h_avg) ----------------
__global__ void global_gates_kernel(const float* __restrict__ g,
                                    const float* __restrict__ havg,
                                    const float* __restrict__ Wgf, const float* __restrict__ bgf,
                                    const float* __restrict__ Wgo, const float* __restrict__ bgo,
                                    float* __restrict__ fg, float* __restrict__ og)
{
    long idx = (long)blockIdx.x * blockDim.x + threadIdx.x;
    if (idx >= (long)CB * CH) return;
    int b  = (int)(idx / CH);
    int hd = (int)(idx % CH);
    float af = bgf[hd];
    float ao = bgo[hd];
    for (int k = 0; k < 2 * CH; ++k) {
        float v = (k < CH) ? g[(long)b * CH + k] : havg[(long)b * CH + (k - CH)];
        af += v * Wgf[(long)k * CH + hd];
        ao += v * Wgo[(long)k * CH + hd];
    }
    fg[idx] = af;
    og[idx] = 1.0f / (1.0f + expf(-ao));
}

// ---------------- pack A for fi GEMM: [g_broadcast | h_new] -> bf16 ----------------
__global__ void pack_afi_kernel(const float* __restrict__ g,
                                const float* __restrict__ hnew,
                                __bf16* __restrict__ afi)
{
    long idx = (long)blockIdx.x * blockDim.x + threadIdx.x;
    if (idx >= (long)CBL * CKFI) return;
    int row = (int)(idx / CKFI);
    int col = (int)(idx % CKFI);
    int b   = row >> 7;
    float v = (col < CH) ? g[(long)b * CH + col]
                         : hnew[(long)row * CH + (col - CH)];
    afi[idx] = (__bf16)v;
}

// ---------------- slot softmax over L+1 slots + cg/g update ----------------
__global__ void slot_update_kernel(const float* __restrict__ fg,
                                   const float* __restrict__ fi,
                                   const float* __restrict__ c_new,
                                   const float* __restrict__ cg,
                                   const float* __restrict__ og,
                                   const int* __restrict__ lengths,
                                   float* __restrict__ cg_new,
                                   float* __restrict__ g_new)
{
    long idx = (long)blockIdx.x * blockDim.x + threadIdx.x;
    if (idx >= (long)CB * CH) return;
    int b  = (int)(idx / CH);
    int hd = (int)(idx % CH);
    int len = lengths[b];

    float mx = fg[idx];
    for (int l = 0; l < CL; ++l)
        if (l < len) mx = fmaxf(mx, fi[(long)(b * CL + l) * CH + hd]);

    float e0 = expf(fg[idx] - mx);
    float s  = e0;
    float wsum = 0.0f;
    for (int l = 0; l < CL; ++l) {
        if (l < len) {
            long o = (long)(b * CL + l) * CH + hd;
            float e = expf(fi[o] - mx);
            s += e;
            wsum += e * c_new[o];
        }
    }
    float inv = 1.0f / s;
    float cgn = e0 * inv * cg[idx] + wsum * inv;
    cg_new[idx] = cgn;
    g_new[idx]  = og[idx] * tanhf(cgn);
}

// ---------------- head: a1 = tanh(g @ W1 + b1) ----------------
__global__ void head1_kernel(const float* __restrict__ g,
                             const float* __restrict__ W1, const float* __restrict__ b1,
                             float* __restrict__ a1)
{
    long idx = (long)blockIdx.x * blockDim.x + threadIdx.x;
    if (idx >= (long)CB * 2 * CH) return;
    int b = (int)(idx / (2 * CH));
    int j = (int)(idx % (2 * CH));
    float acc = b1[j];
    for (int k = 0; k < CH; ++k)
        acc += g[(long)b * CH + k] * W1[(long)k * (2 * CH) + j];
    a1[idx] = tanhf(acc);
}

// ---------------- head: logits + log_softmax ----------------
__global__ void head2_kernel(const float* __restrict__ a1,
                             const float* __restrict__ W2, const float* __restrict__ b2,
                             float* __restrict__ out)
{
    int b = threadIdx.x;
    if (b >= CB) return;
    float lg[CDOUT];
    for (int d = 0; d < CDOUT; ++d) {
        float acc = b2[d];
        for (int k = 0; k < 2 * CH; ++k)
            acc += a1[(long)b * (2 * CH) + k] * W2[(long)k * CDOUT + d];
        lg[d] = acc;
    }
    float mx = lg[0];
    for (int d = 1; d < CDOUT; ++d) mx = fmaxf(mx, lg[d]);
    float s = 0.0f;
    for (int d = 0; d < CDOUT; ++d) s += expf(lg[d] - mx);
    float lse = mx + logf(s);
    for (int d = 0; d < CDOUT; ++d)
        out[(long)b * CDOUT + d] = lg[d] - lse;
}

// ---------------- host orchestration ----------------
static inline dim3 grid1d(long n) { return dim3((unsigned)((n + 255) / 256)); }

extern "C" void kernel_launch(void* const* d_in, const int* in_sizes, int n_in,
                              void* d_out, int out_size, void* d_ws, size_t ws_size,
                              hipStream_t stream)
{
    (void)in_sizes; (void)n_in; (void)out_size; (void)ws_size;

    const int*   tokens  = (const int*)d_in[0];
    const int*   lengths = (const int*)d_in[1];
    const float* embed   = (const float*)d_in[2];
    const float* W0  = (const float*)d_in[3];
    const float* b0  = (const float*)d_in[4];
    const float* Wg  = (const float*)d_in[5];
    const float* bg  = (const float*)d_in[6];
    const float* Wgf = (const float*)d_in[7];
    const float* bgf = (const float*)d_in[8];
    const float* Wfi = (const float*)d_in[9];
    const float* bfi = (const float*)d_in[10];
    const float* Wgo = (const float*)d_in[11];
    const float* bgo = (const float*)d_in[12];
    const float* W1  = (const float*)d_in[13];
    const float* b1  = (const float*)d_in[14];
    const float* W2  = (const float*)d_in[15];
    const float* b2  = (const float*)d_in[16];
    float* out = (float*)d_out;

    char* ws = (char*)d_ws;
    size_t off = 0;
    auto alloc = [&](size_t bytes) -> void* {
        void* p = ws + off;
        off = (off + bytes + 255) & ~(size_t)255;
        return p;
    };

    float*  x     = (float*)  alloc((size_t)CBL * CE  * 4);
    __bf16* xb    = (__bf16*) alloc((size_t)CBL * CK0P * 2);
    __bf16* W0bt  = (__bf16*) alloc((size_t)CH * CK0P * 2);    // [N=CH][K=CK0P]
    __bf16* Wgbt  = (__bf16*) alloc((size_t)CNG * CKGP * 2);   // [N=CNG][K=CKGP]
    __bf16* Wfibt = (__bf16*) alloc((size_t)CH * CKFI * 2);    // [N=CH][K=CKFI]
    float*  hA    = (float*)  alloc((size_t)CBL * CH * 4);
    float*  hBuf  = (float*)  alloc((size_t)CBL * CH * 4);
    float*  cA    = (float*)  alloc((size_t)CBL * CH * 4);
    float*  cBuf  = (float*)  alloc((size_t)CBL * CH * 4);
    float*  gA    = (float*)  alloc((size_t)CB * CH * 4);
    float*  gBuf  = (float*)  alloc((size_t)CB * CH * 4);
    float*  cgA   = (float*)  alloc((size_t)CB * CH * 4);
    float*  cgBuf = (float*)  alloc((size_t)CB * CH * 4);
    __bf16* ctxb  = (__bf16*) alloc((size_t)CBL * CKGP * 2);
    float*  gates = (float*)  alloc((size_t)CBL * CNG * 4);
    __bf16* afib  = (__bf16*) alloc((size_t)CBL * CKFI * 2);
    float*  fi    = (float*)  alloc((size_t)CBL * CH * 4);
    float*  havg  = (float*)  alloc((size_t)CB * CH * 4);
    float*  fg    = (float*)  alloc((size_t)CB * CH * 4);
    float*  og    = (float*)  alloc((size_t)CB * CH * 4);
    float*  a1    = (float*)  alloc((size_t)CB * 2 * CH * 4);

    // --- embedding gather + bf16 pack; weight conversions (transposed, K-padded) ---
    embed_pack_kernel<<<grid1d((long)CBL * CK0P), 256, 0, stream>>>(tokens, embed, x, xb);
    convert_pad_T_kernel<<<grid1d((long)CH * CK0P), 256, 0, stream>>>(W0,  W0bt,  CE,  CK0P, CH);
    convert_pad_T_kernel<<<grid1d((long)CNG * CKGP), 256, 0, stream>>>(Wg,  Wgbt,  CKG, CKGP, CNG);
    convert_pad_T_kernel<<<grid1d((long)CH * CKFI), 256, 0, stream>>>(Wfi, Wfibt, CKFI, CKFI, CH);

    // --- h = tanh(x @ W0 + b0) * mask ; c = 0 ; g = avg(h) ; cg = 0 ---
    {
        dim3 g(CBL / 128, CH / 128);
        gemm_bf16_wmma_kernel<<<g, 256, 0, stream>>>(xb, W0bt, b0, hA,
                                                     CBL, CH, CK0P, 1, lengths);
    }
    hipMemsetAsync(cA,  0, (size_t)CBL * CH * 4, stream);
    hipMemsetAsync(cgA, 0, (size_t)CB  * CH * 4, stream);
    seq_avg_kernel<<<grid1d((long)CB * CH), 256, 0, stream>>>(hA, lengths, gA);

    float *hC = hA, *hN = hBuf, *cC = cA, *cN = cBuf;
    float *gC = gA, *gN = gBuf, *cgC = cgA, *cgN = cgBuf;

    for (int step = 0; step < CNSTEPS; ++step) {
        pack_ctx_kernel<<<grid1d((long)CBL * CKGP), 256, 0, stream>>>(hC, x, gC, ctxb);
        {
            dim3 g(CBL / 128, CNG / 128);
            gemm_bf16_wmma_kernel<<<g, 256, 0, stream>>>(ctxb, Wgbt, bg, gates,
                                                         CBL, CNG, CKGP, 0, lengths);
        }
        cell_update_kernel<<<grid1d((long)CBL * CH), 256, 0, stream>>>(
            gates, cC, cgC, lengths, cN, hN);
        seq_avg_kernel<<<grid1d((long)CB * CH), 256, 0, stream>>>(hN, lengths, havg);
        global_gates_kernel<<<grid1d((long)CB * CH), 256, 0, stream>>>(
            gC, havg, Wgf, bgf, Wgo, bgo, fg, og);
        pack_afi_kernel<<<grid1d((long)CBL * CKFI), 256, 0, stream>>>(gC, hN, afib);
        {
            dim3 g(CBL / 128, CH / 128);
            gemm_bf16_wmma_kernel<<<g, 256, 0, stream>>>(afib, Wfibt, bfi, fi,
                                                         CBL, CH, CKFI, 0, lengths);
        }
        slot_update_kernel<<<grid1d((long)CB * CH), 256, 0, stream>>>(
            fg, fi, cN, cgC, og, lengths, cgN, gN);

        float* t;
        t = hC;  hC  = hN;  hN  = t;
        t = cC;  cC  = cN;  cN  = t;
        t = gC;  gC  = gN;  gN  = t;
        t = cgC; cgC = cgN; cgN = t;
    }

    head1_kernel<<<grid1d((long)CB * 2 * CH), 256, 0, stream>>>(gC, W1, b1, a1);
    head2_kernel<<<1, 32, 0, stream>>>(a1, W2, b2, out);
}
